// SAT1Mamba_25761213841442
// MI455X (gfx1250) — matrix operand order repaired and checked by reading, not compiled, not run on hardware
//
#include <hip/hip_runtime.h>
#include <hip/hip_bf16.h>

// ---------------- types for WMMA ----------------
typedef __attribute__((ext_vector_type(16))) _Float16 v16h;
typedef __attribute__((ext_vector_type(8)))  _Float16 v8h;
typedef __attribute__((ext_vector_type(8)))  float    v8f;

#define BATCH   32
#define LIN     4096
#define LC      4094           // length after conv1/conv2
#define ME      ((size_t)BATCH * LC)   // 131008 rows, multiple of 64
#define DMODEL  128
#define DINNER  256
#define DSTATE  16
#define DTRANK  8
#define NCHUNK  32
#define CHLEN   128            // NCHUNK*CHLEN = 4096 >= LC (last chunk = 126)

// ---------------- workspace layout (bytes) ----------------
// h1 region [0, 0x2000000) is dead after conv2 -> reused for scan chunk partials.
#define OFF_H1     ((size_t)0x0000000)   // ME*64*4   = 33.5 MB  (conv1 out f32)
#define OFF_PCHK   ((size_t)0x0000000)   // 32*32*256*16*4 = 16 MB (chunk products)
#define OFF_SCHK   ((size_t)0x1000000)   // 16 MB (chunk folded sums)
#define OFF_U16    ((size_t)0x2000000)   // ME*128*2  = 33.5 MB  (conv2 out f16, GEMM A)
#define OFF_XI16   ((size_t)0x4000000)   // ME*256*2  = 67 MB    (in_proj xi, f16)
#define OFF_XC16   ((size_t)0x8000000)   // ME*256*2  = 67 MB    (silu(dwconv), f16)
#define OFF_XDBL   ((size_t)0xC000000)   // ME*48*4   = 25 MB    (x_proj out, f32, stride 48)
#define OFF_W16IP  ((size_t)0xE000000)   // 512*128*2
#define OFF_W16XP  ((size_t)0xE020000)   // 48*256*2 (rows 40..47 zeroed)
#define OFF_SILUZ  ((size_t)0xE030000)   // 32*256*4
#define OFF_YLAST  ((size_t)0xE040000)   // 32*256*4

// ---------------- WMMA tile loaders (row-major f16, ISA §7.12.2 layout) ----------------
// A 16x32 (MxK): lane m=lane&15, half=lane>>4; VGPR0-3 = K[half*8 .. +7], VGPR4-7 = +16
__device__ __forceinline__ v16h load_tile_A(const _Float16* base, size_t ld, int mrow, int half, int k0) {
    const _Float16* p = base + (size_t)mrow * ld + half * 8 + k0;
    v8h lo = *(const v8h*)(p);
    v8h hi = *(const v8h*)(p + 16);
    v16h r;
#pragma unroll
    for (int i = 0; i < 8; ++i) { r[i] = lo[i]; r[i + 8] = hi[i]; }
    return r;
}
// B 32x16 (KxN), B[k][n] = W[n][k]: lane n=lane&15, half=lane>>4; VGPR0-3 = K[half*8 .. +7], VGPR4-7 = +16
__device__ __forceinline__ v16h load_tile_B(const _Float16* base, size_t ld, int nrow, int half, int k0) {
    const _Float16* p = base + (size_t)nrow * ld + half * 8 + k0;
    v8h lo = *(const v8h*)(p);
    v8h hi = *(const v8h*)(p + 16);
    v16h r;
#pragma unroll
    for (int i = 0; i < 8; ++i) { r[i] = lo[i]; r[i + 8] = hi[i]; }
    return r;
}

// ---------------- K0: convert weights to f16 (x_proj padded to 48 rows) ----------------
__global__ void cvt_weights(const float* __restrict__ ipw, const float* __restrict__ xpw,
                            _Float16* __restrict__ w16ip, _Float16* __restrict__ w16xp) {
    int i = blockIdx.x * blockDim.x + threadIdx.x;
    if (i < 512 * 128) w16ip[i] = (_Float16)ipw[i];
    if (i < 48 * 256) {
        int r = i >> 8;
        w16xp[i] = (r < 40) ? (_Float16)xpw[i] : (_Float16)0.0f;
    }
}

// ---------------- K1: conv1 (32->64, k=5, pad 1) + ReLU ----------------
__global__ void __launch_bounds__(64) conv1_relu(const float* __restrict__ x,
                                                 const float* __restrict__ w,
                                                 const float* __restrict__ bias,
                                                 float* __restrict__ h1) {
    int bl = blockIdx.x;            // b*LC + l
    int b = bl / LC, l = bl % LC;
    int co = threadIdx.x;           // 0..63
    __shared__ float xw[5 * 32];
    for (int i = threadIdx.x; i < 5 * 32; i += 64) {
        int t = i >> 5, ci = i & 31;
        int pos = l - 1 + t;
        xw[i] = (pos >= 0 && pos < LIN) ? x[((size_t)b * LIN + pos) * 32 + ci] : 0.0f;
    }
    __syncthreads();
    float acc = bias[co];
    const float* wr = w + (size_t)co * 32 * 5;     // (64,32,5)
#pragma unroll
    for (int ci = 0; ci < 32; ++ci)
#pragma unroll
        for (int t = 0; t < 5; ++t)
            acc = fmaf(xw[t * 32 + ci], wr[ci * 5 + t], acc);
    h1[(size_t)bl * 64 + co] = fmaxf(acc, 0.0f);
}

// ---------------- K2: conv2 (64->128, k=3, pad 1) + ReLU -> f16 (GEMM A) ----------------
__global__ void __launch_bounds__(128) conv2_relu(const float* __restrict__ h1,
                                                  const float* __restrict__ w,
                                                  const float* __restrict__ bias,
                                                  _Float16* __restrict__ u16) {
    int bl = blockIdx.x;
    int b = bl / LC, l = bl % LC;
    int co = threadIdx.x;           // 0..127
    __shared__ float hw[3 * 64];
    for (int i = threadIdx.x; i < 3 * 64; i += 128) {
        int t = i >> 6, ci = i & 63;
        int pos = l - 1 + t;
        hw[i] = (pos >= 0 && pos < LC) ? h1[((size_t)b * LC + pos) * 64 + ci] : 0.0f;
    }
    __syncthreads();
    float acc = bias[co];
    const float* wr = w + (size_t)co * 64 * 3;     // (128,64,3)
#pragma unroll
    for (int ci = 0; ci < 64; ++ci)
#pragma unroll
        for (int t = 0; t < 3; ++t)
            acc = fmaf(hw[t * 64 + ci], wr[ci * 3 + t], acc);
    u16[(size_t)bl * 128 + co] = (_Float16)fmaxf(acc, 0.0f);
}

// ---------------- K3: in_proj xi half via WMMA: Xi = U(MEx128) * Wip[0:256]^T -> f16 ----------------
__global__ void __launch_bounds__(128) gemm_inproj(const _Float16* __restrict__ A,
                                                   const _Float16* __restrict__ W,
                                                   _Float16* __restrict__ Xi) {
    int wave = threadIdx.x >> 5, lane = threadIdx.x & 31;
    int m = lane & 15, half = lane >> 4;
    int mbase = blockIdx.x * 64 + wave * 16;
    int nbase = blockIdx.y * 64;
    v8f acc[4] = {};
#pragma unroll
    for (int k0 = 0; k0 < 128; k0 += 32) {
        v16h a = load_tile_A(A, 128, mbase + m, half, k0);
        if (k0 < 96) __builtin_prefetch(A + (size_t)(mbase + m) * 128 + k0 + 32, 0, 3);
#pragma unroll
        for (int nt = 0; nt < 4; ++nt) {
            v16h bm = load_tile_B(W, 128, nbase + nt * 16 + m, half, k0);
            acc[nt] = __builtin_amdgcn_wmma_f32_16x16x32_f16(
                false, a, false, bm, (short)0, acc[nt], false, false);
        }
    }
#pragma unroll
    for (int nt = 0; nt < 4; ++nt)
#pragma unroll
        for (int j = 0; j < 8; ++j)
            Xi[(size_t)(mbase + j + 8 * half) * 256 + nbase + nt * 16 + m] = (_Float16)acc[nt][j];
}

// ---------------- K4: z for last position only + silu (32x256, K=128) ----------------
__global__ void __launch_bounds__(256) zlast_silu(const _Float16* __restrict__ u16,
                                                  const float* __restrict__ ipw,
                                                  float* __restrict__ siluz) {
    int b = blockIdx.x, j = threadIdx.x;
    const _Float16* ur = u16 + ((size_t)b * LC + (LC - 1)) * 128;
    const float* wr = ipw + (size_t)(256 + j) * 128;
    float acc = 0.0f;
#pragma unroll
    for (int d = 0; d < 128; ++d) acc = fmaf((float)ur[d], wr[d], acc);
    siluz[b * 256 + j] = acc / (1.0f + __expf(-acc));
}

// ---------------- K5: depthwise causal conv (k=4, pad 3,0) + silu -> f16 ----------------
__global__ void __launch_bounds__(256) dwconv_silu(const _Float16* __restrict__ xi,
                                                   const float* __restrict__ cw,
                                                   const float* __restrict__ cb,
                                                   _Float16* __restrict__ xc16) {
    int bl = blockIdx.x;
    int b = bl / LC, l = bl % LC;
    int d = threadIdx.x;
    float acc = cb[d];
#pragma unroll
    for (int t = 0; t < 4; ++t) {
        int pos = l - 3 + t;
        if (pos >= 0)
            acc = fmaf((float)xi[((size_t)b * LC + pos) * 256 + d], cw[d * 4 + t], acc);
    }
    float s = acc / (1.0f + __expf(-acc));
    xc16[(size_t)bl * 256 + d] = (_Float16)s;
}

// ---------------- K6: x_proj via WMMA: Xdbl = Xc(MEx256) * Wxp48^T -> f32 stride 48 ----------------
__global__ void __launch_bounds__(128) gemm_xproj(const _Float16* __restrict__ A,
                                                  const _Float16* __restrict__ W,
                                                  float* __restrict__ Xd) {
    int wave = threadIdx.x >> 5, lane = threadIdx.x & 31;
    int m = lane & 15, half = lane >> 4;
    int mbase = blockIdx.x * 64 + wave * 16;
    v8f acc[3] = {};
#pragma unroll
    for (int k0 = 0; k0 < 256; k0 += 32) {
        v16h a = load_tile_A(A, 256, mbase + m, half, k0);
        if (k0 < 224) __builtin_prefetch(A + (size_t)(mbase + m) * 256 + k0 + 32, 0, 3);
#pragma unroll
        for (int nt = 0; nt < 3; ++nt) {
            v16h bm = load_tile_B(W, 256, nt * 16 + m, half, k0);
            acc[nt] = __builtin_amdgcn_wmma_f32_16x16x32_f16(
                false, a, false, bm, (short)0, acc[nt], false, false);
        }
    }
#pragma unroll
    for (int nt = 0; nt < 3; ++nt)
#pragma unroll
        for (int j = 0; j < 8; ++j)
            Xd[(size_t)(mbase + j + 8 * half) * 48 + nt * 16 + m] = acc[nt][j];
}

// ---------------- K7a: chunked scan pass 1 ----------------
// One lane per (b, d, chunk): fold CHLEN steps into P[n] = prod(dA), S[n] = folded input.
// h_after_chunk = P * h_before + S  (exact linear-recurrence decomposition).
__global__ void __launch_bounds__(256) ssm_scan_chunk(const float* __restrict__ xdbl,
                                                      const _Float16* __restrict__ xc16,
                                                      const float* __restrict__ dtw,
                                                      const float* __restrict__ dtb,
                                                      const float* __restrict__ A_log,
                                                      float* __restrict__ Pc,
                                                      float* __restrict__ Sc) {
    int b = blockIdx.x, c = blockIdx.y, d = threadIdx.x;
    int t0 = c * CHLEN;
    int len = (LC - t0 < CHLEN) ? (LC - t0) : CHLEN;
    float wdt[DTRANK];
#pragma unroll
    for (int r = 0; r < DTRANK; ++r) wdt[r] = dtw[d * DTRANK + r];
    float Ar[DSTATE];
#pragma unroll
    for (int n = 0; n < DSTATE; ++n) Ar[n] = -__expf(A_log[d * DSTATE + n]);
    float bias = dtb[d];
    float P[DSTATE], S[DSTATE];
#pragma unroll
    for (int n = 0; n < DSTATE; ++n) { P[n] = 1.0f; S[n] = 0.0f; }
    const float* xrow = xdbl + ((size_t)b * LC + t0) * 48;
    const _Float16* urow = xc16 + ((size_t)b * LC + t0) * 256 + d;
    for (int t = 0; t < len; ++t) {
        const float* xr = xrow + (size_t)t * 48;
        float pre = bias;
#pragma unroll
        for (int r = 0; r < DTRANK; ++r) pre = fmaf(xr[r], wdt[r], pre);
        float delta = (pre > 20.0f) ? pre : log1pf(__expf(pre));
        float du = delta * (float)urow[(size_t)t * 256];
#pragma unroll
        for (int n = 0; n < DSTATE; ++n) {
            float dA = __expf(delta * Ar[n]);
            S[n] = fmaf(dA, S[n], du * xr[8 + n]);
            P[n] *= dA;
        }
    }
    size_t o = (((size_t)b * NCHUNK + c) * 256 + d) * DSTATE;
#pragma unroll
    for (int n = 0; n < DSTATE; ++n) { Pc[o + n] = P[n]; Sc[o + n] = S[n]; }
}

// ---------------- K7b: fold chunks in time order; emit y at t=last ----------------
__global__ void __launch_bounds__(256) ssm_combine(const float* __restrict__ Pc,
                                                   const float* __restrict__ Sc,
                                                   const float* __restrict__ xdbl,
                                                   const _Float16* __restrict__ xc16,
                                                   const float* __restrict__ Dskip,
                                                   const float* __restrict__ siluz,
                                                   float* __restrict__ ylast) {
    int b = blockIdx.x, d = threadIdx.x;
    float h[DSTATE] = {};
    for (int c = 0; c < NCHUNK; ++c) {
        size_t o = (((size_t)b * NCHUNK + c) * 256 + d) * DSTATE;
#pragma unroll
        for (int n = 0; n < DSTATE; ++n)
            h[n] = fmaf(Pc[o + n], h[n], Sc[o + n]);
    }
    const float* xl = xdbl + ((size_t)b * LC + (LC - 1)) * 48;
    float y = 0.0f;
#pragma unroll
    for (int n = 0; n < DSTATE; ++n) y = fmaf(h[n], xl[24 + n], y);
    float ul = (float)xc16[((size_t)b * LC + (LC - 1)) * 256 + d];
    y = (y + ul * Dskip[d]) * siluz[b * 256 + d];
    ylast[b * 256 + d] = y;
}

// ---------------- K8: out_proj (32 rows) + LayerNorm + head ----------------
__global__ void __launch_bounds__(128) head_kernel(const float* __restrict__ ylast,
                                                   const float* __restrict__ opw,
                                                   const float* __restrict__ lng,
                                                   const float* __restrict__ lnb,
                                                   const float* __restrict__ hw,
                                                   const float* __restrict__ hb,
                                                   float* __restrict__ out) {
    int b = blockIdx.x, o = threadIdx.x;     // 128 threads
    __shared__ float red[128];
    __shared__ float norm[128];
    const float* yr = ylast + b * 256;
    const float* wr = opw + (size_t)o * 256;
    float acc = 0.0f;
#pragma unroll
    for (int d = 0; d < 256; ++d) acc = fmaf(yr[d], wr[d], acc);
    red[o] = acc; __syncthreads();
    for (int s = 64; s > 0; s >>= 1) { if (o < s) red[o] += red[o + s]; __syncthreads(); }
    float mu = red[0] / 128.0f; __syncthreads();
    float dev = acc - mu;
    red[o] = dev * dev; __syncthreads();
    for (int s = 64; s > 0; s >>= 1) { if (o < s) red[o] += red[o + s]; __syncthreads(); }
    float var = red[0] / 128.0f;
    norm[o] = dev * rsqrtf(var + 1e-5f) * lng[o] + lnb[o];
    __syncthreads();
    if (o < 5) {
        float a = hb[o];
        const float* h2 = hw + o * 128;
#pragma unroll
        for (int j = 0; j < 128; ++j) a = fmaf(norm[j], h2[j], a);
        out[b * 5 + o] = a;
    }
}

// ---------------- launcher ----------------
extern "C" void kernel_launch(void* const* d_in, const int* in_sizes, int n_in,
                              void* d_out, int out_size, void* d_ws, size_t ws_size,
                              hipStream_t stream) {
    const float* x       = (const float*)d_in[0];
    const float* c1w     = (const float*)d_in[1];
    const float* c1b     = (const float*)d_in[2];
    const float* c2w     = (const float*)d_in[3];
    const float* c2b     = (const float*)d_in[4];
    const float* ipw     = (const float*)d_in[5];
    const float* cdw     = (const float*)d_in[6];
    const float* cdb     = (const float*)d_in[7];
    const float* xpw     = (const float*)d_in[8];
    const float* dtw     = (const float*)d_in[9];
    const float* dtb     = (const float*)d_in[10];
    const float* alog    = (const float*)d_in[11];
    const float* dskip   = (const float*)d_in[12];
    const float* opw     = (const float*)d_in[13];
    const float* lng     = (const float*)d_in[14];
    const float* lnb     = (const float*)d_in[15];
    const float* hw      = (const float*)d_in[16];
    const float* hb      = (const float*)d_in[17];
    float* out           = (float*)d_out;

    char* ws = (char*)d_ws;
    float*     h1    = (float*)    (ws + OFF_H1);
    float*     Pc    = (float*)    (ws + OFF_PCHK);   // reuses dead h1 region
    float*     Sc    = (float*)    (ws + OFF_SCHK);   // reuses dead h1 region
    _Float16*  u16   = (_Float16*) (ws + OFF_U16);
    _Float16*  xi16  = (_Float16*) (ws + OFF_XI16);
    _Float16*  xc16  = (_Float16*) (ws + OFF_XC16);
    float*     xdbl  = (float*)    (ws + OFF_XDBL);
    _Float16*  w16ip = (_Float16*) (ws + OFF_W16IP);
    _Float16*  w16xp = (_Float16*) (ws + OFF_W16XP);
    float*     siluz = (float*)    (ws + OFF_SILUZ);
    float*     ylast = (float*)    (ws + OFF_YLAST);

    const int nBL = (int)(BATCH * LC);         // 131008
    const int mblocks = nBL / 64;              // 2047

    cvt_weights<<<256, 256, 0, stream>>>(ipw, xpw, w16ip, w16xp);
    conv1_relu<<<nBL, 64, 0, stream>>>(x, c1w, c1b, h1);
    conv2_relu<<<nBL, 128, 0, stream>>>(h1, c2w, c2b, u16);
    gemm_inproj<<<dim3(mblocks, 4), 128, 0, stream>>>(u16, w16ip, xi16);
    zlast_silu<<<BATCH, 256, 0, stream>>>(u16, ipw, siluz);
    dwconv_silu<<<nBL, 256, 0, stream>>>(xi16, cdw, cdb, xc16);
    gemm_xproj<<<dim3(mblocks, 1), 128, 0, stream>>>(xc16, w16xp, xdbl);
    ssm_scan_chunk<<<dim3(BATCH, NCHUNK), 256, 0, stream>>>(xdbl, xc16, dtw, dtb, alog, Pc, Sc);
    ssm_combine<<<BATCH, 256, 0, stream>>>(Pc, Sc, xdbl, xc16, dskip, siluz, ylast);
    head_kernel<<<BATCH, 128, 0, stream>>>(ylast, opw, lng, lnb, hw, hb, out);
}
